// HierarchicalAttention_71133248356699
// MI455X (gfx1250) — matrix operand order repaired and checked by reading, not compiled, not run on hardware
//
#include <hip/hip_runtime.h>

// ---------------------------------------------------------------------------
// Types
// ---------------------------------------------------------------------------
typedef __attribute__((ext_vector_type(16))) __bf16 v16bf;
typedef __attribute__((ext_vector_type(8)))  __bf16 v8bf;
typedef __attribute__((ext_vector_type(8)))  float  v8f;

union BF16x16 { v16bf v; v8bf h[2]; };

__device__ __forceinline__ __bf16 f2bf(float f) {
    union { float f; unsigned u; } x; x.f = f;
    unsigned r = (x.u + 0x7FFFu + ((x.u >> 16) & 1u)) >> 16;
    union { unsigned short s; __bf16 b; } y; y.s = (unsigned short)r;
    return y.b;
}

__device__ __forceinline__ v8f wmma_bf16(v16bf a, v16bf b, v8f c) {
    return __builtin_amdgcn_wmma_f32_16x16x32_bf16(false, a, false, b, (short)0, c,
                                                   false, false);
}

// ---------------------------------------------------------------------------
// fp32 -> bf16 conversion
// ---------------------------------------------------------------------------
__global__ void k_cvt(const float* __restrict__ s, __bf16* __restrict__ d, int n) {
    int i = blockIdx.x * blockDim.x + threadIdx.x;
    if (i < n) d[i] = f2bf(s[i]);
}

// ---------------------------------------------------------------------------
// Tiled bf16 GEMM:  C[m,n] = scale * sum_k A[m,k]*Bop[k,n] (+ bias[n])
//   A: row-major [M,K] bf16
//   B: if !B_KXN: weight layout [N,K] (row n contiguous over k)  -> direct stage
//      if  B_KXN: value  layout [K,N]                            -> transposed stage
//   Out: bf16 or fp32, row-major stride N. Batched over blockIdx.z.
//   Block tile 128m x 256n, BLK_K = 32. 8 waves in 2x4 grid; each wave owns a
//   64x64 tile = 16 WMMA accumulators (1.0 ds_load_b128 per v_wmma).
//   Software pipeline: tile kb+1 is prefetched into registers while the WMMAs
//   for tile kb run out of LDS (global latency hidden behind matrix ops).
// ---------------------------------------------------------------------------
template <bool B_KXN, bool OUT_BF16>
__global__ __launch_bounds__(256) void k_gemm(const __bf16* __restrict__ A,
                                              const __bf16* __restrict__ B,
                                              const float* __restrict__ bias,
                                              void* __restrict__ Cv,
                                              int N, int K,
                                              long long aB, long long bB, long long cB,
                                              float scale) {
    __shared__ __bf16 at[128 * 40];   // padded: stride 40 halfs = 80B (16B aligned)
    __shared__ __bf16 bt[256 * 40];   // layout [n][k]

    const int t = threadIdx.x, lane = t & 31, wave = t >> 5;
    const int half = (lane >> 4) & 1, l16 = lane & 15;
    const int nblk = blockIdx.x * 256, mblk = blockIdx.y * 128, z = blockIdx.z;
    const __bf16* Az = A + (size_t)z * (size_t)aB;
    const __bf16* Bz = B + (size_t)z * (size_t)bB;
    const int wm = wave >> 2, wn = wave & 3;   // 2 x 4 wave grid

    v8f acc[4][4] = {};

    // register staging for the software pipeline
    uint4 ra0, ra1;        // A: 2 threads/row, 16 halfs each chunk
    uint4 rb0, rb1;        // B (!B_KXN): 1 thread/row, 16 halfs
    __bf16 rbk[32];        // B (B_KXN): strided gather, 1 thread/col

    const int ar = t >> 1, ac = (t & 1) * 16;

    // preload kb = 0
    {
        const uint4* s = (const uint4*)(Az + (size_t)(mblk + ar) * K + ac);
        ra0 = s[0]; ra1 = s[1];
        if (!B_KXN) {
            const uint4* sb = (const uint4*)(Bz + (size_t)(nblk + t) * K);
            rb0 = sb[0]; rb1 = sb[1];
        } else {
            #pragma unroll
            for (int k = 0; k < 32; ++k)
                rbk[k] = Bz[(size_t)k * N + nblk + t];
        }
    }

    const int ksteps = K >> 5;
    for (int kb = 0; kb < ksteps; ++kb) {
        __syncthreads();
        // commit staged registers to LDS
        *(uint4*)&at[ar * 40 + ac]     = ra0;
        *(uint4*)&at[ar * 40 + ac + 8] = ra1;
        if (!B_KXN) {
            *(uint4*)&bt[t * 40]     = rb0;
            *(uint4*)&bt[t * 40 + 8] = rb1;
        } else {
            #pragma unroll
            for (int k = 0; k < 32; ++k) bt[t * 40 + k] = rbk[k];
        }
        __syncthreads();

        // prefetch tile kb+1 (overlaps with WMMAs below)
        if (kb + 1 < ksteps) {
            const int kc = (kb + 1) * 32;
            const uint4* s = (const uint4*)(Az + (size_t)(mblk + ar) * K + kc + ac);
            ra0 = s[0]; ra1 = s[1];
            if (!B_KXN) {
                const uint4* sb = (const uint4*)(Bz + (size_t)(nblk + t) * K + kc);
                rb0 = sb[0]; rb1 = sb[1];
            } else {
                #pragma unroll
                for (int k = 0; k < 32; ++k)
                    rbk[k] = Bz[(size_t)(kc + k) * N + nblk + t];
            }
        }

        BF16x16 af[4], bfg[4];
        #pragma unroll
        for (int i = 0; i < 4; ++i) {
            int row = wm * 64 + i * 16 + l16;
            const __bf16* p = &at[row * 40 + half * 8];
            af[i].h[0] = *(const v8bf*)p;
            af[i].h[1] = *(const v8bf*)(p + 16);
        }
        #pragma unroll
        for (int j = 0; j < 4; ++j) {
            int n = wn * 64 + j * 16 + l16;
            const __bf16* p = &bt[n * 40 + half * 8];
            bfg[j].h[0] = *(const v8bf*)p;
            bfg[j].h[1] = *(const v8bf*)(p + 16);
        }
        #pragma unroll
        for (int i = 0; i < 4; ++i)
            #pragma unroll
            for (int j = 0; j < 4; ++j)
                acc[i][j] = wmma_bf16(af[i].v, bfg[j].v, acc[i][j]);
    }

    // Epilogue. C layout: VGPR r -> row (r + 8*half), col = lane%16 within tile.
    #pragma unroll
    for (int i = 0; i < 4; ++i)
        #pragma unroll
        for (int j = 0; j < 4; ++j) {
            int n = nblk + wn * 64 + j * 16 + l16;
            float bv = bias ? bias[n] : 0.0f;
            #pragma unroll
            for (int r = 0; r < 8; ++r) {
                int mrow = mblk + wm * 64 + i * 16 + r + 8 * half;
                float v = acc[i][j][r] * scale + bv;
                size_t off = (size_t)z * (size_t)cB + (size_t)mrow * N + n;
                if (OUT_BF16) ((__bf16*)Cv)[off] = f2bf(v);
                else          ((float*)Cv)[off]  = v;
            }
        }
}

// ---------------------------------------------------------------------------
// Local flash attention: per (query-block of 128, head, chunk).
// q,k,v bf16 [16384, 2048]; head h occupies cols h*128..+128.
// 8 waves; wave w owns query rows w*16..+16 (16x128 fp32 accumulator).
// ---------------------------------------------------------------------------
#define FA_STR 136   // 128 + 8 halfs padding (keeps 16B alignment, spreads banks)

__global__ __launch_bounds__(256) void k_flash(const __bf16* __restrict__ Q,
                                               const __bf16* __restrict__ Kg,
                                               const __bf16* __restrict__ Vg,
                                               __bf16* __restrict__ CTX) {
    __shared__ __bf16 qt[128 * FA_STR];   // [qrow][d]
    __shared__ __bf16 kt[128 * FA_STR];   // [key][d]
    __shared__ __bf16 vt[128 * FA_STR];   // [d][key]  (transposed)
    __shared__ __bf16 pl[128 * FA_STR];   // 8 waves x 16 rows of P

    const int qb = blockIdx.x, h = blockIdx.y, ch = blockIdx.z;
    const int t = threadIdx.x, lane = t & 31, wave = t >> 5;
    const int half = (lane >> 4) & 1, l16 = lane & 15;
    const size_t rowBase = (size_t)ch * 1024;
    const int colBase = h * 128;

    // stage Q tile (coalesced b128)
    #pragma unroll
    for (int it = 0; it < 8; ++it) {
        int lin = t + it * 256;              // units of 8 halfs
        int r = lin >> 4, c8 = (lin & 15) << 3;
        const uint4* s = (const uint4*)(Q + (rowBase + qb * 128 + r) * 2048 + colBase + c8);
        *(uint4*)&qt[r * FA_STR + c8] = *s;
    }
    __syncthreads();

    // Q fragments (A operand, 16x32 each, K=d)
    BF16x16 qf[4];
    {
        int row = wave * 16 + l16;
        #pragma unroll
        for (int kk = 0; kk < 4; ++kk) {
            const __bf16* p = &qt[row * FA_STR + kk * 32 + half * 8];
            qf[kk].h[0] = *(const v8bf*)p;
            qf[kk].h[1] = *(const v8bf*)(p + 16);
        }
    }

    v8f O[8] = {};
    float m[8], l[8];
    #pragma unroll
    for (int r = 0; r < 8; ++r) { m[r] = -1e30f; l[r] = 0.0f; }
    const float sc = 0.08838834764831845f;   // 1/sqrt(128)

    for (int kb = 0; kb < 8; ++kb) {
        __syncthreads();
        // stage K tile + transposed V tile
        #pragma unroll
        for (int it = 0; it < 8; ++it) {
            int lin = t + it * 256;
            int r = lin >> 4, c8 = (lin & 15) << 3;
            const uint4* sk = (const uint4*)(Kg + (rowBase + kb * 128 + r) * 2048 + colBase + c8);
            *(uint4*)&kt[r * FA_STR + c8] = *sk;
            union { uint4 u; __bf16 b[8]; } vv;
            vv.u = *(const uint4*)(Vg + (rowBase + kb * 128 + r) * 2048 + colBase + c8);
            #pragma unroll
            for (int j = 0; j < 8; ++j) vt[(c8 + j) * FA_STR + r] = vv.b[j];
        }
        __syncthreads();

        // S = Q * K^T  (B operand column n = key, contiguous over d in kt)
        v8f S[8];
        #pragma unroll
        for (int nt = 0; nt < 8; ++nt) {
            v8f s = {};
            int key = nt * 16 + l16;
            #pragma unroll
            for (int kk = 0; kk < 4; ++kk) {
                BF16x16 bfg;
                const __bf16* p = &kt[key * FA_STR + kk * 32 + half * 8];
                bfg.h[0] = *(const v8bf*)p;
                bfg.h[1] = *(const v8bf*)(p + 16);
                s = wmma_bf16(qf[kk].v, bfg.v, s);
            }
            #pragma unroll
            for (int r = 0; r < 8; ++r) s[r] *= sc;
            S[nt] = s;
        }

        // online softmax update (row = r + 8*half; reduce over the 16-lane half)
        #pragma unroll
        for (int r = 0; r < 8; ++r) {
            float mx = S[0][r];
            #pragma unroll
            for (int nt = 1; nt < 8; ++nt) mx = fmaxf(mx, S[nt][r]);
            #pragma unroll
            for (int msk = 1; msk <= 8; msk <<= 1) mx = fmaxf(mx, __shfl_xor(mx, msk, 32));
            float mn = fmaxf(m[r], mx);
            float c = __expf(m[r] - mn);
            m[r] = mn; l[r] *= c;
            #pragma unroll
            for (int ct = 0; ct < 8; ++ct) O[ct][r] *= c;
            float rs = 0.0f;
            #pragma unroll
            for (int nt = 0; nt < 8; ++nt) {
                float pv = __expf(S[nt][r] - mn);
                S[nt][r] = pv; rs += pv;
            }
            #pragma unroll
            for (int msk = 1; msk <= 8; msk <<= 1) rs += __shfl_xor(rs, msk, 32);
            l[r] += rs;
        }

        // spill P (bf16) through per-wave LDS to reshape into A fragments
        {
            __bf16* pw = &pl[wave * 16 * FA_STR];
            #pragma unroll
            for (int nt = 0; nt < 8; ++nt)
                #pragma unroll
                for (int r = 0; r < 8; ++r)
                    pw[(r + 8 * half) * FA_STR + nt * 16 + l16] = f2bf(S[nt][r]);
        }
        __syncthreads();

        // O += P * V   (B operand column n = d, contiguous over key in vt)
        {
            const __bf16* pw = &pl[wave * 16 * FA_STR];
            #pragma unroll
            for (int kk = 0; kk < 4; ++kk) {
                BF16x16 af;
                const __bf16* pa = &pw[l16 * FA_STR + kk * 32 + half * 8];
                af.h[0] = *(const v8bf*)pa;
                af.h[1] = *(const v8bf*)(pa + 16);
                #pragma unroll
                for (int ct = 0; ct < 8; ++ct) {
                    BF16x16 bfg;
                    int d = ct * 16 + l16;
                    const __bf16* pb = &vt[d * FA_STR + kk * 32 + half * 8];
                    bfg.h[0] = *(const v8bf*)pb;
                    bfg.h[1] = *(const v8bf*)(pb + 16);
                    O[ct] = wmma_bf16(af.v, bfg.v, O[ct]);
                }
            }
        }
    }

    // normalize + store ctx (bf16)
    #pragma unroll
    for (int ct = 0; ct < 8; ++ct)
        #pragma unroll
        for (int r = 0; r < 8; ++r) {
            float o = O[ct][r] / l[r];
            size_t grow = rowBase + qb * 128 + wave * 16 + r + 8 * half;
            int gcol = colBase + ct * 16 + l16;
            CTX[grow * 2048 + gcol] = f2bf(o);
        }
}

// ---------------------------------------------------------------------------
// Row softmax: fp32 [rows,1024] -> bf16 probs
// ---------------------------------------------------------------------------
__global__ __launch_bounds__(256) void k_softmax(const float* __restrict__ S,
                                                 __bf16* __restrict__ P) {
    __shared__ float red[256];
    const int row = blockIdx.x, t = threadIdx.x;
    const float* s = S + (size_t)row * 1024;
    __bf16* p = P + (size_t)row * 1024;
    float v[4], mx = -1e30f;
    #pragma unroll
    for (int j = 0; j < 4; ++j) { v[j] = s[t + j * 256]; mx = fmaxf(mx, v[j]); }
    red[t] = mx; __syncthreads();
    for (int st = 128; st > 0; st >>= 1) { if (t < st) red[t] = fmaxf(red[t], red[t + st]); __syncthreads(); }
    mx = red[0]; __syncthreads();
    float sum = 0.0f;
    #pragma unroll
    for (int j = 0; j < 4; ++j) { v[j] = __expf(v[j] - mx); sum += v[j]; }
    red[t] = sum; __syncthreads();
    for (int st = 128; st > 0; st >>= 1) { if (t < st) red[t] += red[t + st]; __syncthreads(); }
    float inv = 1.0f / red[0];
    #pragma unroll
    for (int j = 0; j < 4; ++j) p[t + j * 256] = f2bf(v[j] * inv);
}

// ---------------------------------------------------------------------------
// Residual + LayerNorm over 2048 cols
// ---------------------------------------------------------------------------
__global__ __launch_bounds__(256) void k_ln(const float* __restrict__ Y0,
                                            const float* __restrict__ X,
                                            const float* __restrict__ g,
                                            const float* __restrict__ b,
                                            float* __restrict__ out) {
    __shared__ float red[256];
    const int row = blockIdx.x, t = threadIdx.x;
    const size_t base = (size_t)row * 2048;
    float y[8], s = 0.0f;
    #pragma unroll
    for (int j = 0; j < 8; ++j) { int c = t + j * 256; y[j] = Y0[base + c] + X[base + c]; s += y[j]; }
    red[t] = s; __syncthreads();
    for (int st = 128; st > 0; st >>= 1) { if (t < st) red[t] += red[t + st]; __syncthreads(); }
    float mean = red[0] * (1.0f / 2048.0f); __syncthreads();
    float s2 = 0.0f;
    #pragma unroll
    for (int j = 0; j < 8; ++j) { float d = y[j] - mean; s2 += d * d; }
    red[t] = s2; __syncthreads();
    for (int st = 128; st > 0; st >>= 1) { if (t < st) red[t] += red[t + st]; __syncthreads(); }
    float var = red[0] * (1.0f / 2048.0f);
    float rinv = rsqrtf(var + 1e-5f);
    #pragma unroll
    for (int j = 0; j < 8; ++j) {
        int c = t + j * 256;
        out[base + c] = (y[j] - mean) * rinv * g[c] + b[c];
    }
}

// ---------------------------------------------------------------------------
// Host launch
// ---------------------------------------------------------------------------
extern "C" void kernel_launch(void* const* d_in, const int* in_sizes, int n_in,
                              void* d_out, int out_size, void* d_ws, size_t ws_size,
                              hipStream_t stream) {
    (void)in_sizes; (void)n_in; (void)out_size; (void)ws_size;
    const int H = 2048;
    const size_t ROWS = 16384;                 // 2 * 8192
    const size_t MATE = (size_t)H * H;         // 4.19M elems per weight
    const size_t XBe  = ROWS * H;              // 33.55M elems per activation

    __bf16* wt   = (__bf16*)d_ws;              // 8 weights bf16        (67.1 MB)
    __bf16* xb   = wt + 8 * MATE;              // x / local bf16        (67.1 MB)
    __bf16* bufA = xb + XBe;                   // q / gq / gctx         (67.1 MB)
    __bf16* bufB = bufA + XBe;                 // k / gk                (67.1 MB)
    __bf16* bufC = bufB + XBe;                 // v / gv                (67.1 MB)
    __bf16* bufD = bufC + XBe;                 // ctx bf16 / gscores f32(67.1 MB)
    __bf16* bufE = bufD + XBe;                 // probs bf16            (33.6 MB)
    float*  outw = (float*)(bufE + XBe / 2);   // op gemm out fp32      (134 MB)
    float*  gsc  = (float*)bufD;

    const float* x = (const float*)d_in[0];

    // 1. weights + x -> bf16
    const int widx[8] = {1, 3, 5, 7, 9, 11, 13, 15};  // lq,lk,lv,lo,gq,gk,gv,op
    for (int i = 0; i < 8; ++i)
        k_cvt<<<dim3((unsigned)((MATE + 255) / 256)), dim3(256), 0, stream>>>(
            (const float*)d_in[widx[i]], wt + i * MATE, (int)MATE);
    k_cvt<<<dim3((unsigned)((XBe + 255) / 256)), dim3(256), 0, stream>>>(x, xb, (int)XBe);

    dim3 gBig(8, 128, 1);   // N=2048 (256-wide tiles), M=16384
    // 2. q,k,v projections
    k_gemm<false, true><<<gBig, 256, 0, stream>>>(xb, wt + 0 * MATE, (const float*)d_in[2],
                                                  bufA, 2048, 2048, 0, 0, 0, 1.0f);
    k_gemm<false, true><<<gBig, 256, 0, stream>>>(xb, wt + 1 * MATE, (const float*)d_in[4],
                                                  bufB, 2048, 2048, 0, 0, 0, 1.0f);
    k_gemm<false, true><<<gBig, 256, 0, stream>>>(xb, wt + 2 * MATE, (const float*)d_in[6],
                                                  bufC, 2048, 2048, 0, 0, 0, 1.0f);
    // 3. local multi-head flash attention -> ctx
    k_flash<<<dim3(8, 16, 16), 256, 0, stream>>>(bufA, bufB, bufC, bufD);
    // 4. local output projection (reuse xb)
    k_gemm<false, true><<<gBig, 256, 0, stream>>>(bufD, wt + 3 * MATE, (const float*)d_in[8],
                                                  xb, 2048, 2048, 0, 0, 0, 1.0f);
    // 5. global q,k,v
    k_gemm<false, true><<<gBig, 256, 0, stream>>>(xb, wt + 4 * MATE, (const float*)d_in[10],
                                                  bufA, 2048, 2048, 0, 0, 0, 1.0f);
    k_gemm<false, true><<<gBig, 256, 0, stream>>>(xb, wt + 5 * MATE, (const float*)d_in[12],
                                                  bufB, 2048, 2048, 0, 0, 0, 1.0f);
    k_gemm<false, true><<<gBig, 256, 0, stream>>>(xb, wt + 6 * MATE, (const float*)d_in[14],
                                                  bufC, 2048, 2048, 0, 0, 0, 1.0f);
    // 6. global scores = gq . gk^T / sqrt(H)   (batched over 16 chunks)
    k_gemm<false, false><<<dim3(4, 8, 16), 256, 0, stream>>>(
        bufA, bufB, nullptr, gsc, 1024, 2048,
        (long long)1024 * 2048, (long long)1024 * 2048, (long long)1024 * 1024,
        0.022097086912079608f);
    // 7. softmax -> bf16 probs
    k_softmax<<<dim3(16384), 256, 0, stream>>>(gsc, bufE);
    // 8. gctx = P . gv    (B is [K=1024, N=2048] row-major -> transposed stage)
    k_gemm<true, true><<<dim3(8, 8, 16), 256, 0, stream>>>(
        bufE, bufC, nullptr, bufA, 2048, 1024,
        (long long)1024 * 1024, (long long)1024 * 2048, (long long)1024 * 2048, 1.0f);
    // 9. output projection (fp32 out)
    k_gemm<false, false><<<gBig, 256, 0, stream>>>(bufA, wt + 7 * MATE,
                                                   (const float*)d_in[16], outw,
                                                   2048, 2048, 0, 0, 0, 1.0f);
    // 10. residual + layernorm
    k_ln<<<dim3(16384), 256, 0, stream>>>(outw, x, (const float*)d_in[17],
                                          (const float*)d_in[18], (float*)d_out);
}
